// MultiHeadAttention_89404039234050
// MI455X (gfx1250) — matrix operand order repaired and checked by reading, not compiled, not run on hardware
//
#include <hip/hip_runtime.h>
#include <hip/hip_bf16.h>

#define B_  2
#define S_  2048
#define D_  768
#define H_  12
#define HD_ 64
#define SCALE_ 0.125f   // 1/sqrt(64)

typedef __attribute__((ext_vector_type(16))) _Float16 v16h;
typedef __attribute__((ext_vector_type(8)))  float    v8f;

typedef int v4i_t __attribute__((vector_size(4 * sizeof(int))));
typedef __attribute__((address_space(1))) v4i_t gv4i_t;   // global int4
typedef __attribute__((address_space(3))) v4i_t lv4i_t;   // LDS int4
typedef __attribute__((address_space(3))) _Float16 lds_f16_t;

__device__ __forceinline__ v8f wmma_f16(v16h a, v16h b, v8f c) {
    // D = A(16x32 f16) * B(32x16 f16) + C(16x16 f32)
    return __builtin_amdgcn_wmma_f32_16x16x32_f16(
        /*neg_a=*/false, a, /*neg_b=*/false, b,
        /*c_mod=*/(short)0, c, /*reuse_a=*/false, /*reuse_b=*/false);
}

union FragU { uint4 u[2]; v16h v; };
union Pack8 { uint4 u; _Float16 h[8]; };

// ---------------------------------------------------------------------------
// Async global->LDS copy (16B per call) + asynccnt wait, with asm fallback
// ---------------------------------------------------------------------------
__device__ __forceinline__ void async_copy_b128(const _Float16* g, _Float16* l) {
#if __has_builtin(__builtin_amdgcn_global_load_async_to_lds_b128)
    __builtin_amdgcn_global_load_async_to_lds_b128(
        (gv4i_t*)g, (lv4i_t*)l, 0, 0);
#else
    unsigned int lofs = (unsigned int)(unsigned long long)(lds_f16_t*)l;
    unsigned long long ga = (unsigned long long)g;
    asm volatile("global_load_async_to_lds_b128 %0, %1, off"
                 :: "v"(lofs), "v"(ga) : "memory");
#endif
}

__device__ __forceinline__ void wait_async0() {
#if __has_builtin(__builtin_amdgcn_s_wait_asynccnt)
    __builtin_amdgcn_s_wait_asynccnt(0);
#else
    asm volatile("s_wait_asynccnt 0" ::: "memory");
#endif
}

// ---------------------------------------------------------------------------
// fp32 -> fp16 conversion
// ---------------------------------------------------------------------------
__global__ __launch_bounds__(256) void cvt_f32_f16(const float* __restrict__ src,
                                                   _Float16* __restrict__ dst, int n) {
    int i = blockIdx.x * blockDim.x + threadIdx.x;
    if (i < n) dst[i] = (_Float16)src[i];
}

// ---------------------------------------------------------------------------
// Panel GEMM: Y = A @ W^T + b.
// Block = 256 threads = 8 waves. Macro-tile 128(M) x 64(N); wave tile 16x64.
// B panel (64 x 32 per k-chunk) staged in LDS via async copies, double-buffered
// so chunk c+1 copy overlaps chunk-c WMMAs; A fragments loaded direct (b128).
// mode 0: Q -> [b,h,s,d] f16, scaled by SCALE_
// mode 1: K -> [b,h,s,d] f16
// mode 2: V -> [b,h,d,s] f16 (transposed, PV B-fragments become contiguous)
// mode 3: fp32 row-major output (final projection)
// ---------------------------------------------------------------------------
__global__ __launch_bounds__(256) void panel_gemm_kernel(
    const _Float16* __restrict__ ah,   // A  [B*S x D]
    const _Float16* __restrict__ wh,   // W  [D x D] (rows = output features)
    const float* __restrict__ bias,
    _Float16* __restrict__ outh,       // modes 0..2
    float* __restrict__ outf,          // mode 3
    int mode) {
    __shared__ _Float16 sB[2][64 * 32];          // 2 x 4 KB panel buffers
    const int tid  = threadIdx.x;
    const int wid  = tid >> 5;
    const int lane = tid & 31;
    const int NB_N = D_ / 64;                    // 12 N-panels
    const int M0 = (blockIdx.x / NB_N) * 128 + wid * 16;
    const int N0 = (blockIdx.x % NB_N) * 64;
    const int cj     = lane & 15;
    const int rbase  = (lane & 16) ? 8 : 0;
    const int ka_off = (lane & 16) ? 16 : 0;     // A-frag K sub-offset
    const int kb_off = (lane & 16) ? 16 : 0;     // B-frag K sub-offset

    const _Float16* aptr = ah + (size_t)(M0 + cj) * D_;

    // this thread's 16B share of each staged panel chunk
    const int sn = tid >> 2;                     // 0..63 : panel row (output feature)
    const int sk = (tid & 3) * 8;                // 0,8,16,24 : k offset in halfs
    const _Float16* gsrc = wh + (size_t)(N0 + sn) * D_ + sk;
    _Float16* l0 = &sB[0][sn * 32 + sk];
    _Float16* l1 = &sB[1][sn * 32 + sk];

    v8f acc[4] = {{}, {}, {}, {}};

    const int NK = D_ / 32;                      // 24 k-chunks
    async_copy_b128(gsrc, l0);
    for (int c = 0; c < NK; ++c) {
        wait_async0();
        __syncthreads();                         // panel c visible to all waves
        if (c + 1 < NK)
            async_copy_b128(gsrc + (c + 1) * 32, ((c + 1) & 1) ? l1 : l0);

        FragU a;
        a.u[0] = *(const uint4*)(aptr + c * 32 + ka_off);
        a.u[1] = *(const uint4*)(aptr + c * 32 + ka_off + 8);
        const _Float16* Bp = &sB[c & 1][0];
        #pragma unroll
        for (int j = 0; j < 4; ++j) {
            FragU b;
            const _Float16* bp = Bp + (j * 16 + cj) * 32 + kb_off;
            b.u[0] = *(const uint4*)(bp);
            b.u[1] = *(const uint4*)(bp + 8);
            acc[j] = wmma_f16(a.v, b.v, acc[j]);
        }
    }

    if (mode == 3) {
        #pragma unroll
        for (int j = 0; j < 4; ++j) {
            const int n = N0 + j * 16 + cj;
            const float bn = bias[n];
            #pragma unroll
            for (int r = 0; r < 8; ++r)
                outf[(size_t)(M0 + rbase + r) * D_ + n] = acc[j][r] + bn;
        }
    } else if (mode == 2) {
        // V transposed: 8 consecutive s per lane -> one b128 store per tile
        const int bb = M0 >> 11;
        const int sb = (M0 & (S_ - 1)) + rbase;
        #pragma unroll
        for (int j = 0; j < 4; ++j) {
            const int n = N0 + j * 16 + cj;
            const float bn = bias[n];
            const int h = n >> 6, d = n & 63;
            Pack8 pk;
            #pragma unroll
            for (int r = 0; r < 8; ++r) pk.h[r] = (_Float16)(acc[j][r] + bn);
            *(uint4*)(outh + ((size_t)(bb * H_ + h) * HD_ + d) * S_ + sb) = pk.u;
        }
    } else {
        const float sc = (mode == 0) ? SCALE_ : 1.0f;
        #pragma unroll
        for (int j = 0; j < 4; ++j) {
            const int n = N0 + j * 16 + cj;
            const float bn = bias[n];
            const int h = n >> 6, d = n & 63;
            #pragma unroll
            for (int r = 0; r < 8; ++r) {
                const int m = M0 + rbase + r;
                const int bb = m >> 11, s = m & (S_ - 1);
                outh[((size_t)(bb * H_ + h) * S_ + s) * HD_ + d] =
                    (_Float16)((acc[j][r] + bn) * sc);
            }
        }
    }
}

// ---------------------------------------------------------------------------
// Flash attention: one wave = one (b,h,q-tile of 16). Streams 32-key chunks.
// ---------------------------------------------------------------------------
__global__ __launch_bounds__(256) void attention_kernel(const _Float16* __restrict__ Qh,
                                                        const _Float16* __restrict__ Kh,
                                                        const _Float16* __restrict__ Vt,
                                                        _Float16* __restrict__ At) {
    __shared__ _Float16 sP[8][16 * 32];          // per-wave 16x32 P tile (8 KB)
    const int wid  = threadIdx.x >> 5;
    const int lane = threadIdx.x & 31;
    const int task = blockIdx.x * 8 + wid;
    if (task >= B_ * H_ * (S_ / 16)) return;
    const int qt = task & ((S_ / 16) - 1);       // 0..127
    const int bh = task >> 7;
    const int b  = bh / H_, h = bh % H_;
    const int cj     = lane & 15;
    const int rbase  = (lane & 16) ? 8 : 0;
    const int ka_off = (lane & 16) ? 16 : 0;
    const int kb_off = (lane & 16) ? 16 : 0;
    const int q0 = qt * 16;

    // load Q A-fragments once (16 q-rows x 64 dims -> 2 fragments)
    const _Float16* qptr = Qh + ((size_t)bh * S_ + q0 + cj) * HD_;
    v16h aq0, aq1;
    {
        FragU t;
        t.u[0] = *(const uint4*)(qptr + ka_off);
        t.u[1] = *(const uint4*)(qptr + ka_off + 8);
        aq0 = t.v;
        t.u[0] = *(const uint4*)(qptr + 32 + ka_off);
        t.u[1] = *(const uint4*)(qptr + 32 + ka_off + 8);
        aq1 = t.v;
    }

    v8f acc0 = {}, acc1 = {}, acc2 = {}, acc3 = {};
    float m[8], l[8];
    #pragma unroll
    for (int r = 0; r < 8; ++r) { m[r] = -3.0e38f; l[r] = 0.0f; }

    const _Float16* Kbase = Kh + (size_t)bh * S_ * HD_;
    const _Float16* Vbase = Vt + (size_t)bh * HD_ * S_;
    _Float16* Pw = sP[wid];

    const int nch = (q0 + 16 + 31) >> 5;         // causal chunk count
    for (int c = 0; c < nch; ++c) {
        const int kc0 = c * 32;

        // --- scores: Q(16x64) @ K^T(64x32) -> two 16x16 f32 tiles -----------
        v8f s0 = {}, s1 = {};
        {
            FragU t;
            const _Float16* kp0 = Kbase + (size_t)(kc0 + cj) * HD_;
            t.u[0] = *(const uint4*)(kp0 + kb_off);
            t.u[1] = *(const uint4*)(kp0 + kb_off + 8);
            s0 = wmma_f16(aq0, t.v, s0);
            t.u[0] = *(const uint4*)(kp0 + 32 + kb_off);
            t.u[1] = *(const uint4*)(kp0 + 32 + kb_off + 8);
            s0 = wmma_f16(aq1, t.v, s0);
            const _Float16* kp1 = Kbase + (size_t)(kc0 + 16 + cj) * HD_;
            t.u[0] = *(const uint4*)(kp1 + kb_off);
            t.u[1] = *(const uint4*)(kp1 + kb_off + 8);
            s1 = wmma_f16(aq0, t.v, s1);
            t.u[0] = *(const uint4*)(kp1 + 32 + kb_off);
            t.u[1] = *(const uint4*)(kp1 + 32 + kb_off + 8);
            s1 = wmma_f16(aq1, t.v, s1);
        }

        // --- causal mask + online softmax ----------------------------------
        #pragma unroll
        for (int r = 0; r < 8; ++r) {
            const int qrow = q0 + rbase + r;
            if (kc0 + cj      > qrow) s0[r] = -3.0e38f;
            if (kc0 + 16 + cj > qrow) s1[r] = -3.0e38f;

            float rm = fmaxf(s0[r], s1[r]);
            rm = fmaxf(rm, __shfl_xor(rm, 1, 32));
            rm = fmaxf(rm, __shfl_xor(rm, 2, 32));
            rm = fmaxf(rm, __shfl_xor(rm, 4, 32));
            rm = fmaxf(rm, __shfl_xor(rm, 8, 32));
            const float mnew  = fmaxf(m[r], rm);
            const float scale = expf(m[r] - mnew);
            const float p0 = expf(s0[r] - mnew);
            const float p1 = expf(s1[r] - mnew);
            float rs = p0 + p1;
            rs += __shfl_xor(rs, 1, 32);
            rs += __shfl_xor(rs, 2, 32);
            rs += __shfl_xor(rs, 4, 32);
            rs += __shfl_xor(rs, 8, 32);
            l[r] = l[r] * scale + rs;
            m[r] = mnew;
            acc0[r] *= scale; acc1[r] *= scale;
            acc2[r] *= scale; acc3[r] *= scale;
            // C-layout -> LDS (row q, col key), f16
            Pw[(rbase + r) * 32 + cj]      = (_Float16)p0;
            Pw[(rbase + r) * 32 + cj + 16] = (_Float16)p1;
        }
        asm volatile("s_wait_dscnt 0" ::: "memory");

        // --- re-layout P as A-fragment (16x32) -----------------------------
        v16h pa;
        {
            FragU t;
            const _Float16* pr = Pw + (lane & 15) * 32;
            t.u[0] = *(const uint4*)(pr + ka_off);
            t.u[1] = *(const uint4*)(pr + 8 + ka_off);
            pa = t.v;
        }

        // --- PV: P(16x32) @ V(32x64), V transposed -> contiguous loads -----
        {
            FragU t;
            const _Float16* vp = Vbase + (size_t)cj * S_ + kc0 + kb_off;
            t.u[0] = *(const uint4*)(vp);
            t.u[1] = *(const uint4*)(vp + 8);
            acc0 = wmma_f16(pa, t.v, acc0);
            vp += (size_t)16 * S_;
            t.u[0] = *(const uint4*)(vp);
            t.u[1] = *(const uint4*)(vp + 8);
            acc1 = wmma_f16(pa, t.v, acc1);
            vp += (size_t)16 * S_;
            t.u[0] = *(const uint4*)(vp);
            t.u[1] = *(const uint4*)(vp + 8);
            acc2 = wmma_f16(pa, t.v, acc2);
            vp += (size_t)16 * S_;
            t.u[0] = *(const uint4*)(vp);
            t.u[1] = *(const uint4*)(vp + 8);
            acc3 = wmma_f16(pa, t.v, acc3);
        }
    }

    // --- finalize: divide by l, store f16 in [b, s, h*64+d] -----------------
    #pragma unroll
    for (int r = 0; r < 8; ++r) {
        const float inv = 1.0f / l[r];
        const int srow = q0 + rbase + r;
        _Float16* orow = At + ((size_t)b * S_ + srow) * D_ + h * HD_;
        orow[cj]      = (_Float16)(acc0[r] * inv);
        orow[16 + cj] = (_Float16)(acc1[r] * inv);
        orow[32 + cj] = (_Float16)(acc2[r] * inv);
        orow[48 + cj] = (_Float16)(acc3[r] * inv);
    }
}

// ---------------------------------------------------------------------------
extern "C" void kernel_launch(void* const* d_in, const int* in_sizes, int n_in,
                              void* d_out, int out_size, void* d_ws, size_t ws_size,
                              hipStream_t stream) {
    (void)in_sizes; (void)n_in; (void)out_size; (void)ws_size;
    const float* x  = (const float*)d_in[0];
    const float* Wq = (const float*)d_in[1];
    const float* bq = (const float*)d_in[2];
    const float* Wk = (const float*)d_in[3];
    const float* bk = (const float*)d_in[4];
    const float* Wv = (const float*)d_in[5];
    const float* bv = (const float*)d_in[6];
    const float* Wo = (const float*)d_in[7];
    const float* bo = (const float*)d_in[8];
    float* out = (float*)d_out;

    const size_t NX = (size_t)B_ * S_ * D_;   // 3,145,728
    const size_t NW = (size_t)D_ * D_;        // 589,824
    _Float16* xh  = (_Float16*)d_ws;
    _Float16* wqh = xh  + NX;
    _Float16* wkh = wqh + NW;
    _Float16* wvh = wkh + NW;
    _Float16* woh = wvh + NW;
    _Float16* Qh  = woh + NW;
    _Float16* Kh  = Qh  + NX;
    _Float16* Vt  = Kh  + NX;
    _Float16* At  = Vt  + NX;                 // total ~36.2 MB of workspace

    const int thr = 256;
    cvt_f32_f16<<<(int)((NX + thr - 1) / thr), thr, 0, stream>>>(x,  xh,  (int)NX);
    cvt_f32_f16<<<(int)((NW + thr - 1) / thr), thr, 0, stream>>>(Wq, wqh, (int)NW);
    cvt_f32_f16<<<(int)((NW + thr - 1) / thr), thr, 0, stream>>>(Wk, wkh, (int)NW);
    cvt_f32_f16<<<(int)((NW + thr - 1) / thr), thr, 0, stream>>>(Wv, wvh, (int)NW);
    cvt_f32_f16<<<(int)((NW + thr - 1) / thr), thr, 0, stream>>>(Wo, woh, (int)NW);

    // 128x64 macro-tiles: (4096/128) * (768/64) = 384 blocks
    const int blocksG = ((B_ * S_) / 128) * (D_ / 64);
    panel_gemm_kernel<<<blocksG, 256, 0, stream>>>(xh, wqh, bq, Qh, nullptr, 0);
    panel_gemm_kernel<<<blocksG, 256, 0, stream>>>(xh, wkh, bk, Kh, nullptr, 1);
    panel_gemm_kernel<<<blocksG, 256, 0, stream>>>(xh, wvh, bv, Vt, nullptr, 2);

    const int tasksB = B_ * H_ * (S_ / 16);          // 3072 wave-tasks
    attention_kernel<<<tasksB / 8, 256, 0, stream>>>(Qh, Kh, Vt, At);

    panel_gemm_kernel<<<blocksG, 256, 0, stream>>>(At, woh, bo, nullptr, out, 3);
}
